// Model_36163624632545
// MI455X (gfx1250) — matrix operand order repaired and checked by reading, not compiled, not run on hardware
//
#include <hip/hip_runtime.h>
#include <hip/hip_bf16.h>

typedef __attribute__((ext_vector_type(16))) _Float16 v16h;
typedef __attribute__((ext_vector_type(8)))  _Float16 v8h;
typedef __attribute__((ext_vector_type(8)))  float    v8f;

#define BB 8
#define NN 20000
#define UU 512
#define CC 64
#define NEG_SLOPE 0.2f
#define MASK_C 500.0f

// ---------- helpers ----------
__device__ __forceinline__ int lbound(const int* __restrict__ a, int n, int val) {
    int lo = 0, hi = n;
    while (lo < hi) { int mid = (lo + hi) >> 1; if (a[mid] < val) lo = mid + 1; else hi = mid; }
    return lo;
}
// monotonic float->uint key so atomicMax(uint) == float max; key(x) > 0 for all reals
__device__ __forceinline__ unsigned fkey(float f) {
    int i = __float_as_int(f); unsigned u = (unsigned)i;
    return (i >= 0) ? (u | 0x80000000u) : ~u;
}
__device__ __forceinline__ float fkeyinv(unsigned u) {
    int i = (u & 0x80000000u) ? (int)(u & 0x7fffffffu) : (int)~u;
    return __int_as_float(i);
}

// ---------- agg[b][u][c] = segment_sum of x over incidences of hyperedge u ----------
__global__ __launch_bounds__(256) void k_agg(const float* __restrict__ x,
                                             const int* __restrict__ nodes,
                                             const int* __restrict__ edges, int E,
                                             float* __restrict__ agg) {
    int u = blockIdx.x;
    int start = lbound(edges, E, u), end = lbound(edges, E, u + 1);
    int t = threadIdx.x;
    int b0 = t >> 6, c0 = t & 63, b1 = b0 + 4;
    float a0 = 0.f, a1 = 0.f;
    for (int e = start; e < end; ++e) {
        int nd = nodes[e];
        a0 += x[((size_t)b0 * NN + nd) * CC + c0];
        a1 += x[((size_t)b1 * NN + nd) * CC + c0];
    }
    agg[((size_t)b0 * UU + u) * CC + c0] = a0;
    agg[((size_t)b1 * UU + u) * CC + c0] = a1;
}

// ---------- node degree ----------
__global__ void k_deg(const int* __restrict__ nodes, int E, unsigned* __restrict__ degD) {
    int e = blockIdx.x * blockDim.x + threadIdx.x;
    if (e < E) atomicAdd(&degD[nodes[e]], 1u);
}

// ---------- xatt[b][n] = dot(x[b,n,:], att[0:64]) ----------
__global__ void k_xatt(const float* __restrict__ x, const float* __restrict__ att,
                       float* __restrict__ xatt) {
    int idx = blockIdx.x * blockDim.x + threadIdx.x;
    if (idx >= BB * NN) return;
    int b = idx / NN, n = idx % NN;
    const float4* xr = (const float4*)(x + ((size_t)b * NN + n) * CC);
    float s = 0.f;
#pragma unroll
    for (int j = 0; j < 16; ++j) {
        float4 xv = xr[j];
        s += xv.x * att[j * 4 + 0] + xv.y * att[j * 4 + 1] +
             xv.z * att[j * 4 + 2] + xv.w * att[j * 4 + 3];
    }
    xatt[idx] = s;
}

// ---------- QKV projection: one wave computes a 16x64 tile of q/k/v ----------
__global__ __launch_bounds__(32) void k_qkv(const float* __restrict__ agg,
    const float* __restrict__ Wq, const float* __restrict__ bq,
    const float* __restrict__ Wk, const float* __restrict__ bk,
    const float* __restrict__ Wv, const float* __restrict__ bv,
    _Float16* __restrict__ qh, _Float16* __restrict__ kh, _Float16* __restrict__ vth) {
    int lane = threadIdx.x, lm = lane & 15, hi = lane >> 4;
    int mtile = blockIdx.x, which = blockIdx.y, b = blockIdx.z;
    const float* W    = (which == 0) ? Wq : (which == 1) ? Wk : Wv;
    const float* bias = (which == 0) ? bq : (which == 1) ? bk : bv;
    const float* arow = agg + ((size_t)b * UU + mtile * 16 + lm) * CC;
    v16h af[2];
#pragma unroll
    for (int ki = 0; ki < 2; ++ki) {
        int kb = ki * 32;
#pragma unroll
        for (int j = 0; j < 8; ++j) {
            af[ki][j]     = (_Float16)arow[kb + hi * 16 + j];
            af[ki][j + 8] = (_Float16)arow[kb + hi * 16 + 8 + j];
        }
    }
#pragma unroll
    for (int nt = 0; nt < 4; ++nt) {
        int n = nt * 16 + lm;
        v8f acc = {};
#pragma unroll
        for (int ki = 0; ki < 2; ++ki) {
            const float* wrow = W + (size_t)n * CC + ki * 32 + hi * 16;
            v16h bf;
#pragma unroll
            for (int j = 0; j < 16; ++j) bf[j] = (_Float16)wrow[j];
            acc = __builtin_amdgcn_wmma_f32_16x16x32_f16(false, af[ki], false, bf,
                                                         (short)0, acc, false, false);
        }
        float bbv = bias[n];
        if (which < 2) {
            _Float16* dst = ((which == 0) ? qh : kh) + (size_t)b * UU * CC;
#pragma unroll
            for (int v = 0; v < 8; ++v) {
                int mo = mtile * 16 + v + 8 * hi;
                dst[(size_t)mo * CC + n] = (_Float16)(acc[v] + bbv);
            }
        } else {
            v8h pk;
#pragma unroll
            for (int v = 0; v < 8; ++v) pk[v] = (_Float16)(acc[v] + bbv);
            _Float16* dst = vth + (size_t)b * CC * UU + (size_t)n * UU + mtile * 16 + 8 * hi;
            *(v8h*)dst = pk;
        }
    }
}

// ---------- attention: per (b, 16-row tile): QK^T -> mask -> softmax -> @V -> eratt ----------
__global__ __launch_bounds__(32) void k_attn(const _Float16* __restrict__ qh,
                                             const _Float16* __restrict__ kh,
                                             const _Float16* __restrict__ vth,
                                             const float* __restrict__ adj,
                                             const float* __restrict__ att,
                                             float* __restrict__ eratt) {
    __shared__ float    sc[16][513];
    __shared__ _Float16 pr[16][520];
    __shared__ float    invs[16];
    __shared__ float    erp[16][33];
    int lane = threadIdx.x, lm = lane & 15, hi = lane >> 4;
    int mtile = blockIdx.x, b = blockIdx.y;

    const _Float16* qrow = qh + ((size_t)b * UU + mtile * 16 + lm) * CC;
    v16h af[2];
#pragma unroll
    for (int ki = 0; ki < 2; ++ki) {
        int kb = ki * 32;
#pragma unroll
        for (int j = 0; j < 8; ++j) {
            af[ki][j]     = qrow[kb + hi * 16 + j];
            af[ki][j + 8] = qrow[kb + hi * 16 + 8 + j];
        }
    }
    for (int nt = 0; nt < 32; ++nt) {
        v8f acc = {};
#pragma unroll
        for (int ki = 0; ki < 2; ++ki) {
            const _Float16* krow = kh + ((size_t)b * UU + nt * 16 + lm) * CC + ki * 32 + hi * 16;
            v16h bf = *(const v16h*)krow;
            acc = __builtin_amdgcn_wmma_f32_16x16x32_f16(false, af[ki], false, bf,
                                                         (short)0, acc, false, false);
        }
#pragma unroll
        for (int v = 0; v < 8; ++v) {
            int um = mtile * 16 + v + 8 * hi;
            int un = nt * 16 + lm;
            sc[v + 8 * hi][un] = acc[v] * 0.125f - (1.0f - adj[(size_t)um * UU + un]) * MASK_C;
        }
    }
    __syncthreads();
    {   // softmax: lane handles half-row (row = lm, cols [hi*256, hi*256+256))
        int row = lm, base = hi * 256;
        float mx = -3.0e38f;
        for (int c = 0; c < 256; ++c) mx = fmaxf(mx, sc[row][base + c]);
        mx = fmaxf(mx, __shfl_xor(mx, 16, 32));
        float sum = 0.f;
        for (int c = 0; c < 256; ++c) {
            float e = __expf(sc[row][base + c] - mx);
            pr[row][base + c] = (_Float16)e;
            sum += e;
        }
        sum += __shfl_xor(sum, 16, 32);
        if (hi == 0) invs[row] = 1.0f / sum;
    }
    __syncthreads();
    float ep[8];
#pragma unroll
    for (int v = 0; v < 8; ++v) ep[v] = 0.f;
    for (int ntc = 0; ntc < 4; ++ntc) {
        v8f acc = {};
        for (int ki = 0; ki < 16; ++ki) {
            int kb = ki * 32;
            v16h afp;
#pragma unroll
            for (int j = 0; j < 8; ++j) {
                afp[j]     = pr[lm][kb + hi * 16 + j];
                afp[j + 8] = pr[lm][kb + hi * 16 + 8 + j];
            }
            const _Float16* vrow = vth + ((size_t)b * CC + ntc * 16 + lm) * UU + kb + hi * 16;
            v16h bf = *(const v16h*)vrow;
            acc = __builtin_amdgcn_wmma_f32_16x16x32_f16(false, afp, false, bf,
                                                         (short)0, acc, false, false);
        }
#pragma unroll
        for (int v = 0; v < 8; ++v) {
            int ml = v + 8 * hi;
            int c  = ntc * 16 + lm;
            ep[v] += acc[v] * invs[ml] * att[CC + c];
        }
    }
#pragma unroll
    for (int v = 0; v < 8; ++v) erp[v + 8 * hi][lane] = ep[v];
    __syncthreads();
    if (lane < 16) {
        int row = lane; float s = 0.f; int c0 = (row < 8) ? 0 : 16;
#pragma unroll
        for (int j = 0; j < 16; ++j) s += erp[row][c0 + j];
        eratt[(size_t)(mtile * 16 + row) * BB + b] = s;
    }
}

// ---------- alpha pass 1: raw leaky-relu'd logits + per-node max ----------
__global__ void k_alpha1(const int* __restrict__ nodes, const int* __restrict__ edges, int E,
                         const float* __restrict__ xatt, const float* __restrict__ eratt,
                         float* __restrict__ raw, unsigned* __restrict__ amax) {
    int idx = blockIdx.x * blockDim.x + threadIdx.x;
    if (idx >= E * BB) return;
    int e = idx >> 3, b = idx & 7;
    int nd = nodes[e], u = edges[e];
    float a = xatt[(size_t)b * NN + nd] + eratt[(size_t)u * BB + b];
    a = (a >= 0.f) ? a : NEG_SLOPE * a;
    raw[idx] = a;
    atomicMax(&amax[(size_t)nd * BB + b], fkey(a));
}

// ---------- alpha pass 2: exp(a - max), per-node denom ----------
__global__ void k_alpha2(const int* __restrict__ nodes, int E,
                         float* __restrict__ raw, const unsigned* __restrict__ amax,
                         float* __restrict__ denom) {
    int idx = blockIdx.x * blockDim.x + threadIdx.x;
    if (idx >= E * BB) return;
    int e = idx >> 3, b = idx & 7;
    int nd = nodes[e];
    float mx = fkeyinv(amax[(size_t)nd * BB + b]);
    float ev = __expf(raw[idx] - mx);
    raw[idx] = ev;
    atomicAdd(&denom[(size_t)nd * BB + b], ev);
}

// ---------- stage 1: nodes -> hyperedges (segment sum, no atomics) ----------
__global__ __launch_bounds__(256) void k_out1(const float* __restrict__ x,
                                              const int* __restrict__ nodes,
                                              const int* __restrict__ edges, int E,
                                              const float* __restrict__ ex,
                                              const float* __restrict__ denom,
                                              float* __restrict__ out1) {
    int u = blockIdx.x;
    int start = lbound(edges, E, u), end = lbound(edges, E, u + 1);
    float Bn = (end > start) ? 1.0f / (float)(end - start) : 0.0f;
    int t = threadIdx.x;
    int b0 = t >> 6, c0 = t & 63, b1 = b0 + 4;
    float a0 = 0.f, a1 = 0.f;
    for (int e = start; e < end; ++e) {
        int nd = nodes[e];
        float w0 = ex[(size_t)e * BB + b0] / (denom[(size_t)nd * BB + b0] + 1e-16f);
        float w1 = ex[(size_t)e * BB + b1] / (denom[(size_t)nd * BB + b1] + 1e-16f);
        a0 += x[((size_t)b0 * NN + nd) * CC + c0] * w0;
        a1 += x[((size_t)b1 * NN + nd) * CC + c0] * w1;
    }
    out1[(size_t)u * BB * CC + b0 * CC + c0] = a0 * Bn;
    out1[(size_t)u * BB * CC + b1 * CC + c0] = a1 * Bn;
}

// ---------- stage 2: hyperedges -> nodes (atomic scatter into d_out) ----------
__global__ __launch_bounds__(256) void k_out2(const int* __restrict__ nodes,
                                              const int* __restrict__ edges, int E,
                                              const float* __restrict__ ex,
                                              const float* __restrict__ denom,
                                              const unsigned* __restrict__ degD,
                                              const float* __restrict__ out1,
                                              float* __restrict__ out) {
    int e = blockIdx.x;
    if (e >= E) return;
    int u = edges[e], nd = nodes[e];
    float Dn = (float)degD[nd];
    int t = threadIdx.x;
    int b0 = t >> 6, c0 = t & 63, b1 = b0 + 4;
    float w0 = ex[(size_t)e * BB + b0] / (denom[(size_t)nd * BB + b0] + 1e-16f) * Dn;
    float w1 = ex[(size_t)e * BB + b1] / (denom[(size_t)nd * BB + b1] + 1e-16f) * Dn;
    atomicAdd(&out[((size_t)b0 * NN + nd) * CC + c0], out1[(size_t)u * BB * CC + b0 * CC + c0] * w0);
    atomicAdd(&out[((size_t)b1 * NN + nd) * CC + c0], out1[(size_t)u * BB * CC + b1 * CC + c0] * w1);
}

extern "C" void kernel_launch(void* const* d_in, const int* in_sizes, int n_in,
                              void* d_out, int out_size, void* d_ws, size_t ws_size,
                              hipStream_t stream) {
    const float* x   = (const float*)d_in[0];
    const float* Wq  = (const float*)d_in[1];
    const float* bq  = (const float*)d_in[2];
    const float* Wk  = (const float*)d_in[3];
    const float* bk  = (const float*)d_in[4];
    const float* Wv  = (const float*)d_in[5];
    const float* bv  = (const float*)d_in[6];
    const float* att = (const float*)d_in[7];
    const float* adj = (const float*)d_in[8];
    const int*   hei = (const int*)d_in[9];
    int E = in_sizes[9] / 2;
    const int* nodes = hei;
    const int* edges = hei + E;

    char* w = (char*)d_ws;
    auto alloc = [&](size_t bytes) -> char* {
        char* p = w; w += (bytes + 255) & ~(size_t)255; return p;
    };
    float*     agg   = (float*)alloc(sizeof(float) * BB * UU * CC);
    _Float16*  qh    = (_Float16*)alloc(sizeof(_Float16) * BB * UU * CC);
    _Float16*  kh    = (_Float16*)alloc(sizeof(_Float16) * BB * UU * CC);
    _Float16*  vth   = (_Float16*)alloc(sizeof(_Float16) * BB * UU * CC);
    float*     eratt = (float*)alloc(sizeof(float) * UU * BB);
    float*     xatt  = (float*)alloc(sizeof(float) * BB * NN);
    float*     exb   = (float*)alloc(sizeof(float) * (size_t)E * BB);
    unsigned*  amax  = (unsigned*)alloc(sizeof(unsigned) * NN * BB);
    float*     denom = (float*)alloc(sizeof(float) * NN * BB);
    unsigned*  degD  = (unsigned*)alloc(sizeof(unsigned) * NN);
    float*     out1  = (float*)alloc(sizeof(float) * UU * BB * CC);

    hipMemsetAsync(d_out, 0, sizeof(float) * (size_t)out_size, stream);
    hipMemsetAsync(amax,  0, sizeof(unsigned) * NN * BB, stream);
    hipMemsetAsync(denom, 0, sizeof(float) * NN * BB, stream);
    hipMemsetAsync(degD,  0, sizeof(unsigned) * NN, stream);

    k_agg<<<UU, 256, 0, stream>>>(x, nodes, edges, E, agg);
    k_deg<<<(E + 255) / 256, 256, 0, stream>>>(nodes, E, degD);
    k_xatt<<<(BB * NN + 255) / 256, 256, 0, stream>>>(x, att, xatt);
    k_qkv<<<dim3(UU / 16, 3, BB), 32, 0, stream>>>(agg, Wq, bq, Wk, bk, Wv, bv, qh, kh, vth);
    k_attn<<<dim3(UU / 16, BB), 32, 0, stream>>>(qh, kh, vth, adj, att, eratt);
    int nab = (E * BB + 255) / 256;
    k_alpha1<<<nab, 256, 0, stream>>>(nodes, edges, E, xatt, eratt, exb, amax);
    k_alpha2<<<nab, 256, 0, stream>>>(nodes, E, exb, amax, denom);
    k_out1<<<UU, 256, 0, stream>>>(x, nodes, edges, E, exb, denom, out1);
    k_out2<<<E, 256, 0, stream>>>(nodes, edges, E, exb, denom, degD, out1, (float*)d_out);
}
